// PyramidCrossAttention_76106820485744
// MI455X (gfx1250) — matrix operand order repaired and checked by reading, not compiled, not run on hardware
//
#include <hip/hip_runtime.h>

typedef __attribute__((ext_vector_type(16))) _Float16 v16h;
typedef __attribute__((ext_vector_type(8)))  _Float16 v8h;
typedef __attribute__((ext_vector_type(8)))  float    v8f;
typedef __attribute__((ext_vector_type(4)))  float    v4f;

#define NKV    4
#define CCH    128
#define NHEADS 4
#define DH     32
#define IMGW   256
#define IMGHW  65536

// K-index for A-matrix (16-bit, 16x32) element j (0..15) given lane-half hi.
// ISA layout: VGPR0-3 -> K 0..7 (lanes<16) / 8..15 (lanes>=16); VGPR4-7 -> K 16..23 / 24..31.
__device__ __forceinline__ int a_kmap(int j, int hi) {
  return (j < 8) ? (j + 8 * hi) : (j + 8 + 8 * hi);
}

__device__ __forceinline__ v8f wmma16(v16h a, v16h b, v8f c) {
  return __builtin_amdgcn_wmma_f32_16x16x32_f16(false, a, false, b, (short)0, c,
                                                false, false);
}

// Assemble A-tile from two contiguous 8-half chunks (matches a_kmap layout).
__device__ __forceinline__ v16h a_from_chunks(const _Float16* base, int hi) {
  v8h lo = *(const v8h*)(base + 8 * hi);
  v8h hh = *(const v8h*)(base + 16 + 8 * hi);
  return __builtin_shufflevector(lo, hh, 0, 1, 2, 3, 4, 5, 6, 7,
                                 8, 9, 10, 11, 12, 13, 14, 15);
}

// ---------------- stage 0: channel means + gate MLP -----------------------
__global__ void k_mean(const float* __restrict__ q, float* __restrict__ gmean) {
  __shared__ float red[256];
  int c = blockIdx.x, t = threadIdx.x;
  float s = 0.f;
  for (int i = t; i < IMGHW; i += 256) s += q[c * IMGHW + i];
  red[t] = s;
  __syncthreads();
  for (int off = 128; off; off >>= 1) {
    if (t < off) red[t] += red[t + off];
    __syncthreads();
  }
  if (t == 0) gmean[c] = red[0] * (1.0f / IMGHW);
}

__global__ void k_gate(const float* __restrict__ gmean,
                       const float* __restrict__ w1, const float* __restrict__ b1,
                       const float* __restrict__ w2, const float* __restrict__ b2,
                       float* __restrict__ gate) {
  __shared__ float gm[CCH], hid[32];
  int t = threadIdx.x;
  gm[t] = gmean[t];
  __syncthreads();
  if (t < 32) {
    float s = b1[t];
    for (int c = 0; c < CCH; ++c) s += w1[t * CCH + c] * gm[c];
    hid[t] = fmaxf(s, 0.f);
  }
  __syncthreads();
  float s = b2[t];
  for (int h = 0; h < 32; ++h) s += w2[t * 32 + h] * hid[h];
  gate[t] = 1.f / (1.f + __expf(-s));
}

// ------------- stage 1: grouped 3x3 conv (WMMA implicit GEMM) -------------
__global__ void k_lowconv(const float* __restrict__ q, const float* __restrict__ lfw,
                          const float* __restrict__ lfb, const float* __restrict__ gate,
                          float* __restrict__ low) {
  int lane = threadIdx.x, hi = lane >> 4, m = lane & 15;
  int o0 = blockIdx.y * 16;
  int grp = blockIdx.y >> 2;              // groups=2, 64 in-ch each
  int y = blockIdx.x >> 4, x0 = (blockIdx.x & 15) << 4;
  v8f acc = {};
  for (int t = 0; t < 9; ++t) {
    int dy = t / 3 - 1, dx = t % 3 - 1;
    int yy = y + dy;
    if (yy < 0 || yy >= IMGW) continue;
    for (int c0 = 0; c0 < 64; c0 += 32) {
      v16h a, b;
#pragma unroll
      for (int j = 0; j < 16; ++j)
        a[j] = (_Float16)lfw[((o0 + m) * 64 + c0 + a_kmap(j, hi)) * 9 + t];
      const float* qrow = q + (size_t)(grp * 64 + c0) * IMGHW + yy * IMGW;
#pragma unroll
      for (int j = 0; j < 16; ++j) {
        int xx = x0 + (lane & 15) + dx;
        int xc = xx < 0 ? 0 : (xx > IMGW - 1 ? IMGW - 1 : xx);
        float v = qrow[(size_t)a_kmap(j, hi) * IMGHW + xc];  // unconditional (clamped) load
        b[j] = (_Float16)((xx == xc) ? v : 0.f);  // branch-free boundary mask
      }
      acc = wmma16(a, b, acc);
    }
  }
  int x = x0 + (lane & 15);
  for (int r = 0; r < 8; ++r) {
    int o = o0 + r + 8 * hi;
    low[(size_t)o * IMGHW + y * IMGW + x] = (acc[r] + lfb[o]) * gate[o];
  }
}

// ---------------- stage 2: exact average pooling --------------------------
__global__ void k_pool(const float* __restrict__ in, float* __restrict__ out,
                       int Cn, int Hin, int f) {
  int Hout = Hin / f;
  int total = Cn * Hout * Hout;
  int idx = blockIdx.x * 256 + threadIdx.x;
  if (idx >= total) return;
  int xo = idx % Hout, yo = (idx / Hout) % Hout, c = idx / (Hout * Hout);
  const float* p = in + ((size_t)c * Hin + yo * f) * Hin + xo * f;
  float s = 0.f;
  for (int a = 0; a < f; ++a)
    for (int b = 0; b < f; ++b) s += p[a * Hin + b];
  out[idx] = s / (float)(f * f);
}

// ---------------- stage 3: 1x1 projection GEMM (WMMA) ---------------------
__global__ void k_proj(const float* __restrict__ W, const float* __restrict__ bias,
                       const float* __restrict__ X, _Float16* __restrict__ O,
                       int L, float scale) {
  int lane = threadIdx.x, hi = lane >> 4, m = lane & 15;
  int l0 = blockIdx.x * 16, o0 = blockIdx.y * 16;
  const float* Xn = X + (size_t)blockIdx.z * CCH * L;
  _Float16* On = O + (size_t)blockIdx.z * CCH * L;
  v8f acc = {};
  for (int c0 = 0; c0 < CCH; c0 += 32) {
    v16h a, b;
#pragma unroll
    for (int j = 0; j < 16; ++j)
      a[j] = (_Float16)W[(o0 + m) * CCH + c0 + a_kmap(j, hi)];
    const float* xrow = Xn + (size_t)c0 * L + l0 + (lane & 15);
#pragma unroll
    for (int j = 0; j < 16; ++j)
      b[j] = (_Float16)xrow[(size_t)a_kmap(j, hi) * L];
    acc = wmma16(a, b, acc);
  }
  for (int r = 0; r < 8; ++r) {
    int o = o0 + r + 8 * hi;
    On[(size_t)o * L + l0 + (lane & 15)] = (_Float16)((acc[r] + bias[o]) * scale);
  }
}

// ---------------- stage 4: multi-head cross attention (WMMA) --------------
__global__ void k_attn(const _Float16* __restrict__ qh,
                       const _Float16* __restrict__ kh,
                       const _Float16* __restrict__ vh,
                       _Float16* __restrict__ cmb, int L) {
  __shared__ float sc[16][1024];
  int lane = threadIdx.x, hi = lane >> 4, m = lane & 15, nn = lane & 15;
  int l0 = blockIdx.x * 16, h = blockIdx.y;

  v16h qa;                              // q tile: rows l0+m, K=d (pre-scaled by d^-1/2)
#pragma unroll
  for (int j = 0; j < 16; ++j)
    qa[j] = qh[(size_t)(h * DH + a_kmap(j, hi)) * L + (l0 + m)];

  v8f acc0 = {}, acc1 = {};
  for (int n = 0; n < NKV; ++n) {
    const _Float16* kn = kh + (size_t)(n * CCH + h * DH) * L;
    for (int ct = 0; ct < L / 16; ++ct) {      // scores = q k^T
      v16h b;
#pragma unroll
      for (int j = 0; j < 16; ++j)
        b[j] = kn[(size_t)a_kmap(j, hi) * L + ct * 16 + nn];
      v8f s = {};
      s = wmma16(qa, b, s);
      for (int r = 0; r < 8; ++r) sc[r + 8 * hi][ct * 16 + nn] = s[r];
    }
    for (int mm = 0; mm < 16; ++mm) {          // wave softmax per row
      float mx = -3.0e38f;
      for (int i = lane; i < L; i += 32) mx = fmaxf(mx, sc[mm][i]);
      for (int off = 16; off; off >>= 1) mx = fmaxf(mx, __shfl_xor(mx, off, 32));
      float sum = 0.f;
      for (int i = lane; i < L; i += 32) {
        float e = __expf(sc[mm][i] - mx);
        sc[mm][i] = e;
        sum += e;
      }
      for (int off = 16; off; off >>= 1) sum += __shfl_xor(sum, off, 32);
      float inv = 1.f / sum;
      for (int i = lane; i < L; i += 32) sc[mm][i] *= inv;
    }
    const _Float16* vn = vh + (size_t)(n * CCH + h * DH) * L;
    for (int k0 = 0; k0 < L; k0 += 32) {       // o += P * V
      v16h a, b0, b1;
      const float* p1 = &sc[m][k0 + 8 * hi];
      const float* p2 = &sc[m][k0 + 16 + 8 * hi];
#pragma unroll
      for (int j = 0; j < 8; ++j) {
        a[j] = (_Float16)p1[j];
        a[j + 8] = (_Float16)p2[j];
      }
#pragma unroll
      for (int j = 0; j < 16; ++j) {
        b0[j] = vn[(size_t)nn * L + k0 + a_kmap(j, hi)];
        b1[j] = vn[(size_t)(16 + nn) * L + k0 + a_kmap(j, hi)];
      }
      acc0 = wmma16(a, b0, acc0);
      acc1 = wmma16(a, b1, acc1);
    }
  }
  for (int r = 0; r < 8; ++r) {                // mean over N=4, store [C][L]
    int l = l0 + r + 8 * hi;
    cmb[(size_t)(h * DH + nn) * L + l] = (_Float16)(0.25f * acc0[r]);
    cmb[(size_t)(h * DH + 16 + nn) * L + l] = (_Float16)(0.25f * acc1[r]);
  }
}

// ----------- stage 5: fold fuse-1x1 into refine-3x3 weights ---------------
__global__ void k_foldw(const float* __restrict__ fuse_w,
                        const float* __restrict__ refine_w,
                        const float* __restrict__ fw3, _Float16* __restrict__ E) {
  int idx = blockIdx.x * 256 + threadIdx.x;
  if (idx >= 3 * CCH * 9 * CCH) return;
  int c = idx & 127, t = (idx >> 7) % 9;
  int o = (idx / (9 * CCH)) % CCH, i = idx / (CCH * 9 * CCH);
  float e0 = __expf(fw3[0]), e1 = __expf(fw3[1]), e2 = __expf(fw3[2]);
  float es = e0 + e1 + e2;
  float wi = ((i == 0) ? e0 : (i == 1) ? e1 : e2) / es;
  float s = 0.f;
  for (int cp = 0; cp < CCH; ++cp)
    s += fuse_w[o * 384 + i * CCH + cp] * refine_w[(cp * CCH + c) * 9 + t];
  E[((size_t)(i * CCH + o)) * 1152 + t * CCH + c] = (_Float16)(wi * s);
}

__global__ void k_cbias(const float* __restrict__ fuse_w, const float* __restrict__ fuse_b,
                        const float* __restrict__ refine_b, const float* __restrict__ fw3,
                        float* __restrict__ cb) {
  int o = threadIdx.x;
  float e0 = __expf(fw3[0]), e1 = __expf(fw3[1]), e2 = __expf(fw3[2]);
  float es = e0 + e1 + e2;
  float wsm0 = e0 / es, wsm1 = e1 / es, wsm2 = e2 / es;
  float s = fuse_b[o];
  for (int i = 0; i < 3; ++i) {
    float t = 0.f;
    for (int c = 0; c < CCH; ++c) t += fuse_w[o * 384 + i * CCH + c] * refine_b[c];
    s += ((i == 0) ? wsm0 : (i == 1) ? wsm1 : wsm2) * t;
  }
  cb[o] = s;
}

// -------- stage 6: fused refine+fuse conv over upsampled pyramids ---------
__global__ void k_fuseconv(const _Float16* __restrict__ E,
                           const _Float16* __restrict__ cm0,
                           const _Float16* __restrict__ cm1,
                           const _Float16* __restrict__ cm2,
                           const float* __restrict__ query,
                           const float* __restrict__ low,
                           const float* __restrict__ cb,
                           const float* __restrict__ hfs,
                           float* __restrict__ out) {
  int lane = threadIdx.x, hi = lane >> 4, m = lane & 15;
  int o0 = blockIdx.y * 16;
  int y = blockIdx.x >> 4, x0 = (blockIdx.x & 15) << 4;
  const _Float16* cms[3] = {cm0, cm1, cm2};
  v8f acc = {};
  for (int i = 0; i < 3; ++i) {
    const _Float16* cm = cms[i];
    int shift = 5 - i, Ws = IMGW >> shift, Lsz = Ws * Ws;
    for (int t = 0; t < 9; ++t) {
      int dy = t / 3 - 1, dx = t % 3 - 1;
      int yy = y + dy;
      if (yy < 0 || yy >= IMGW) continue;
      int ys = yy >> shift;
      const _Float16* Ea = E + (size_t)(i * CCH + o0 + m) * 1152 + t * CCH;
      if (t < 8) __builtin_prefetch(Ea + CCH, 0, 3);
      const _Float16* crow = cm + ys * Ws;
      for (int c0 = 0; c0 < CCH; c0 += 32) {
        v16h a = a_from_chunks(Ea + c0, hi);    // two 16B vector loads
        v16h b;
        const _Float16* cl = crow + (size_t)c0 * Lsz;
#pragma unroll
        for (int j = 0; j < 16; ++j) {
          int xx = x0 + (lane & 15) + dx;
          int xc = xx < 0 ? 0 : (xx > IMGW - 1 ? IMGW - 1 : xx);
          _Float16 v = cl[(size_t)a_kmap(j, hi) * Lsz + (xc >> shift)];  // unconditional (clamped) load
          b[j] = (xx == xc) ? v : (_Float16)0.f; // branch-free boundary mask
        }
        acc = wmma16(a, b, acc);
      }
    }
  }
  int x = x0 + (lane & 15);
  int p = y * IMGW + x;
  float hf = hfs[0];
  for (int r = 0; r < 8; ++r) {
    int o = o0 + r + 8 * hi;
    out[(size_t)o * IMGHW + p] =
        acc[r] + cb[o] + hf * (query[(size_t)o * IMGHW + p] - low[(size_t)o * IMGHW + p]);
  }
}

// ---------------------------------------------------------------------------
extern "C" void kernel_launch(void* const* d_in, const int* in_sizes, int n_in,
                              void* d_out, int out_size, void* d_ws, size_t ws_size,
                              hipStream_t stream) {
  (void)in_sizes; (void)n_in; (void)out_size;
  const float* query  = (const float*)d_in[0];
  const float* keys   = (const float*)d_in[1];
  const float* values = (const float*)d_in[2];
  const float* fw_w1  = (const float*)d_in[3];
  const float* fw_b1  = (const float*)d_in[4];
  const float* fw_w2  = (const float*)d_in[5];
  const float* fw_b2  = (const float*)d_in[6];
  const float* lf_w   = (const float*)d_in[7];
  const float* lf_b   = (const float*)d_in[8];
  // d_in[9..26]: attn params in SCALES order (32,16,8), each qw,qb,kw,kb,vw,vb
  const float* refine_w = (const float*)d_in[27];
  const float* refine_b = (const float*)d_in[28];
  const float* fw3      = (const float*)d_in[29];
  const float* fuse3_w  = (const float*)d_in[30];
  const float* fuse3_b  = (const float*)d_in[31];
  const float* hf_scale = (const float*)d_in[32];
  float* out = (float*)d_out;

  const int Ls[3] = {64, 256, 1024};   // i=0 -> s=32, i=1 -> s=16, i=2 -> s=8

  char* p = (char*)d_ws;
  auto alloc = [&](size_t b) -> void* {
    void* r = (void*)p;
    p += (b + 255) & ~(size_t)255;
    return r;
  };
  float* low   = (float*)alloc((size_t)CCH * IMGHW * 4);
  float* gmean = (float*)alloc(512);
  float* gate  = (float*)alloc(512);
  float* cb    = (float*)alloc(512);
  float *qp[3], *kp[3], *vp[3];
  _Float16 *qh[3], *kh[3], *vh[3], *cmb[3];
  for (int i = 0; i < 3; ++i) {
    size_t L = (size_t)Ls[i];
    qp[i]  = (float*)alloc(CCH * L * 4);
    kp[i]  = (float*)alloc((size_t)NKV * CCH * L * 4);
    vp[i]  = (float*)alloc((size_t)NKV * CCH * L * 4);
    qh[i]  = (_Float16*)alloc(CCH * L * 2);
    kh[i]  = (_Float16*)alloc((size_t)NKV * CCH * L * 2);
    vh[i]  = (_Float16*)alloc((size_t)NKV * CCH * L * 2);
    cmb[i] = (_Float16*)alloc(CCH * L * 2);
  }
  _Float16* E = (_Float16*)alloc((size_t)3 * CCH * 1152 * 2);
  if ((size_t)(p - (char*)d_ws) > ws_size) return;   // ws too small: no-op

  k_mean<<<CCH, 256, 0, stream>>>(query, gmean);
  k_gate<<<1, CCH, 0, stream>>>(gmean, fw_w1, fw_b1, fw_w2, fw_b2, gate);
  k_lowconv<<<dim3(4096, 8), 32, 0, stream>>>(query, lf_w, lf_b, gate, low);

  auto blocks = [](int total) { return (total + 255) / 256; };
  // nested pooling: /8 once, then /2, /2
  k_pool<<<blocks(CCH * 1024), 256, 0, stream>>>(low, qp[2], CCH, IMGW, 8);
  k_pool<<<blocks(CCH * 256), 256, 0, stream>>>(qp[2], qp[1], CCH, 32, 2);
  k_pool<<<blocks(CCH * 64), 256, 0, stream>>>(qp[1], qp[0], CCH, 16, 2);
  k_pool<<<blocks(NKV * CCH * 1024), 256, 0, stream>>>(keys, kp[2], NKV * CCH, IMGW, 8);
  k_pool<<<blocks(NKV * CCH * 256), 256, 0, stream>>>(kp[2], kp[1], NKV * CCH, 32, 2);
  k_pool<<<blocks(NKV * CCH * 64), 256, 0, stream>>>(kp[1], kp[0], NKV * CCH, 16, 2);
  k_pool<<<blocks(NKV * CCH * 1024), 256, 0, stream>>>(values, vp[2], NKV * CCH, IMGW, 8);
  k_pool<<<blocks(NKV * CCH * 256), 256, 0, stream>>>(vp[2], vp[1], NKV * CCH, 32, 2);
  k_pool<<<blocks(NKV * CCH * 64), 256, 0, stream>>>(vp[1], vp[0], NKV * CCH, 16, 2);

  const float qscale = 0.17677669529663687f;  // 1/sqrt(d), folded into q proj
  for (int i = 0; i < 3; ++i) {
    int L = Ls[i];
    int base = 9 + i * 6;
    const float* qw = (const float*)d_in[base + 0];
    const float* qb = (const float*)d_in[base + 1];
    const float* kw = (const float*)d_in[base + 2];
    const float* kb = (const float*)d_in[base + 3];
    const float* vw = (const float*)d_in[base + 4];
    const float* vb = (const float*)d_in[base + 5];
    k_proj<<<dim3(L / 16, 8, 1), 32, 0, stream>>>(qw, qb, qp[i], qh[i], L, qscale);
    k_proj<<<dim3(L / 16, 8, NKV), 32, 0, stream>>>(kw, kb, kp[i], kh[i], L, 1.0f);
    k_proj<<<dim3(L / 16, 8, NKV), 32, 0, stream>>>(vw, vb, vp[i], vh[i], L, 1.0f);
    k_attn<<<dim3(L / 16, NHEADS), 32, 0, stream>>>(qh[i], kh[i], vh[i], cmb[i], L);
  }

  k_foldw<<<(3 * CCH * 9 * CCH + 255) / 256, 256, 0, stream>>>(fuse3_w, refine_w, fw3, E);
  k_cbias<<<1, CCH, 0, stream>>>(fuse3_w, fuse3_b, refine_b, fw3, cb);
  k_fuseconv<<<dim3(4096, 8), 32, 0, stream>>>(E, cmb[0], cmb[1], cmb[2], query, low,
                                               cb, hf_scale, out);
}
